// ConvFormer_75411035783208
// MI455X (gfx1250) — compile-verified, matrix operand order checked by reading, not run on hardware
//
#include <hip/hip_runtime.h>
#include <math.h>

typedef unsigned short ushort_t;
typedef __attribute__((ext_vector_type(16))) __bf16 v16bf;
typedef __attribute__((ext_vector_type(8)))  float  v8f;
typedef __attribute__((ext_vector_type(4)))  int    vint4_t;

#define DEVI __device__ __forceinline__

#if __has_builtin(__builtin_amdgcn_global_load_async_to_lds_b128) && \
    __has_builtin(__builtin_amdgcn_s_wait_asynccnt)
#define USE_ASYNC_LDS 1
#else
#define USE_ASYNC_LDS 0
#endif

namespace {

constexpr int Bn = 8, Ln = 4096, Cn = 512, Hn = 2048;
constexpr int BLn = Bn * Ln;
constexpr float EPSV = 1e-5f;

union Frag16 { v16bf v; unsigned u[8]; };
union Pack8 { uint4 v; ushort_t s[8]; };

#if USE_ASYNC_LDS
typedef __attribute__((address_space(1))) vint4_t* g4_t;
typedef __attribute__((address_space(3))) vint4_t* l4_t;
// generic->LDS: low 32 bits of a generic shared pointer are the LDS offset
DEVI void async_cp16(void* lds, const void* g) {
  __builtin_amdgcn_global_load_async_to_lds_b128(
      (g4_t)(unsigned long long)g,
      (l4_t)(unsigned)(unsigned long long)lds, 0, 0);
}
DEVI void async_wait0() { __builtin_amdgcn_s_wait_asynccnt(0); }
#endif

DEVI ushort_t f2bf(float f) {
  unsigned u = __builtin_bit_cast(unsigned, f);
  unsigned r = u + 0x7FFFu + ((u >> 16) & 1u);   // round-to-nearest-even
  return (ushort_t)(r >> 16);
}

// 16x32 bf16 fragment from an LDS row (K contiguous). Per lane, K={0..7}+8*hi
// and K={16..23}+8*hi are two contiguous 16B runs -> two ds_load_b128.
DEVI void load_frag(Frag16& f, const ushort_t* base, int hi) {
  const uint4* p = (const uint4*)(base + (hi << 3));
  uint4 lo = p[0];
  uint4 hh = p[2];
  f.u[0] = lo.x; f.u[1] = lo.y; f.u[2] = lo.z; f.u[3] = lo.w;
  f.u[4] = hh.x; f.u[5] = hh.y; f.u[6] = hh.z; f.u[7] = hh.w;
}

DEVI v8f wmma_bf16(const Frag16& a, const Frag16& b, v8f c) {
  return __builtin_amdgcn_wmma_f32_16x16x32_bf16(false, a.v, false, b.v,
                                                 (short)0, c, false, false);
}

// ---------------- prep kernels ----------------

__global__ __launch_bounds__(256)
void cvt_bf16_kernel(const float* __restrict__ src, ushort_t* __restrict__ dst,
                     int total8) {
  int idx = blockIdx.x * 256 + threadIdx.x;
  if (idx >= total8) return;
  const float4* s = (const float4*)src + (size_t)idx * 2;
  float4 a = s[0], b = s[1];
  Pack8 o;
  o.s[0] = f2bf(a.x); o.s[1] = f2bf(a.y); o.s[2] = f2bf(a.z); o.s[3] = f2bf(a.w);
  o.s[4] = f2bf(b.x); o.s[5] = f2bf(b.y); o.s[6] = f2bf(b.z); o.s[7] = f2bf(b.w);
  ((uint4*)dst)[idx] = o.v;
}

// W[co][ci][k] (f32) -> Wb[k][co][ci] (bf16)
__global__ void conv_w_reorder_kernel(const float* __restrict__ W,
                                      ushort_t* __restrict__ Wb, int Ktap, int total) {
  int idx = blockIdx.x * blockDim.x + threadIdx.x;
  if (idx >= total) return;
  int k = idx % Ktap;
  int rem = idx / Ktap;
  int ci = rem % Cn;
  int co = rem / Cn;
  Wb[(size_t)k * Cn * Cn + (size_t)co * Cn + ci] = f2bf(W[idx]);
}

// src[R][Ccols] f32 -> dst[Ccols][R] bf16 (GEMM B becomes K-contiguous)
__global__ __launch_bounds__(256)
void wtrans_kernel(const float* __restrict__ src, ushort_t* __restrict__ dst,
                   int R, int Ccols) {
  __shared__ float tile[32][33];
  const int r0 = blockIdx.y * 32, c0 = blockIdx.x * 32;
  const int tx = threadIdx.x, ty = threadIdx.y;  // 32 x 8
#pragma unroll
  for (int jj = 0; jj < 4; ++jj)
    tile[ty + jj * 8][tx] = src[(size_t)(r0 + ty + jj * 8) * Ccols + c0 + tx];
  __syncthreads();
#pragma unroll
  for (int jj = 0; jj < 4; ++jj)
    dst[(size_t)(c0 + ty + jj * 8) * R + r0 + tx] = f2bf(tile[tx][ty + jj * 8]);
}

// ---------------- masked conv as WMMA GEMM ----------------
// Activations stay [B, L, C] bf16 (cin-contiguous == WMMA K-contiguous).
// block tile: 128 (co) x 64 (pos), 8 waves, each wave 32x32 (4 accumulators).

template <int KTAP, bool LAST>
__global__ __launch_bounds__(256)
void conv_kernel(const ushort_t* __restrict__ Xsrc, const ushort_t* __restrict__ Wk,
                 const float* __restrict__ bias, const int* __restrict__ ids,
                 ushort_t* __restrict__ Ybf, float* __restrict__ Yf) {
  constexpr int PL = (KTAP - 1) / 2;
  constexpr int BCOLS = 64 + KTAP - 1;
  __shared__ alignas(16) ushort_t Alds[KTAP][128][32];  // per-tap W tile [co][ci]
  __shared__ alignas(16) ushort_t Blds[BCOLS][32];      // input tile [pos][ci]

  const int b = blockIdx.z;
  const int c0 = blockIdx.y * 128;
  const int i0 = blockIdx.x * 64;
  const int tid = threadIdx.x;
  const int lane = tid & 31, lanelo = lane & 15, hi = lane >> 4;
  const int wave = tid >> 5, mi = wave & 3, npair = wave >> 2;

  // per-lane (per output column) mask bit per tap
  const int* idrow = ids + (size_t)b * Ln;
  unsigned mbits[2] = {0u, 0u};
#pragma unroll
  for (int t = 0; t < 2; ++t) {
    int pc = i0 + npair * 32 + t * 16 + lanelo;
    int idc = idrow[pc];
#pragma unroll
    for (int k = 0; k < KTAP; ++k) {
      int p = pc + k - PL;
      if (p >= 0 && p < Ln && idrow[p] == idc) mbits[t] |= 1u << k;
    }
  }

  v8f acc[2][2];
#pragma unroll
  for (int m = 0; m < 2; ++m)
#pragma unroll
    for (int t = 0; t < 2; ++t)
#pragma unroll
      for (int r = 0; r < 8; ++r) acc[m][t][r] = 0.f;

  const int qa = tid & 3, r0a = tid >> 2;  // q*8 ushort offset, rows r0a / r0a+64
  const ushort_t* wgbase = Wk + (size_t)(c0 + r0a) * Cn + qa * 8;
  const ushort_t* Xb_b = Xsrc + (size_t)b * Ln * Cn;
  const bool interior = (i0 - PL >= 0) && (i0 + 63 + (KTAP - 1 - PL) < Ln);

  for (int cb = 0; cb < Cn / 32; ++cb) {
    // ---- A tiles (always fully in-range) ----
#if USE_ASYNC_LDS
#pragma unroll
    for (int it = 0; it < 2 * KTAP; ++it) {
      const int k = it >> 1;
      const int rofs = (it & 1) * 64;
      async_cp16(&Alds[k][r0a + rofs][qa * 8],
                 wgbase + (size_t)k * Cn * Cn + (size_t)rofs * Cn + cb * 32);
    }
#else
#pragma unroll
    for (int it = 0; it < 2 * KTAP; ++it) {
      const int k = it >> 1;
      const int rofs = (it & 1) * 64;
      uint4 w = *(const uint4*)(wgbase + (size_t)k * Cn * Cn + (size_t)rofs * Cn + cb * 32);
      *(uint4*)&Alds[k][r0a + rofs][qa * 8] = w;
    }
#endif
    // ---- B tile: [pos][ci], ci contiguous in memory ----
    if (interior) {
      for (int idx = tid; idx < BCOLS * 4; idx += 256) {
        int col = idx >> 2, q = idx & 3;
        const ushort_t* src = Xb_b + (size_t)(i0 + col - PL) * Cn + cb * 32 + q * 8;
#if USE_ASYNC_LDS
        async_cp16(&Blds[col][q * 8], src);
#else
        *(uint4*)&Blds[col][q * 8] = *(const uint4*)src;
#endif
      }
    } else {
      for (int idx = tid; idx < BCOLS * 4; idx += 256) {
        int col = idx >> 2, q = idx & 3;
        int pos = i0 + col - PL;
        uint4 v = {0u, 0u, 0u, 0u};
        if (pos >= 0 && pos < Ln)
          v = *(const uint4*)(Xb_b + (size_t)pos * Cn + cb * 32 + q * 8);
        *(uint4*)&Blds[col][q * 8] = v;
      }
    }
#if USE_ASYNC_LDS
    async_wait0();
#endif
    __syncthreads();

#pragma unroll
    for (int k = 0; k < KTAP; ++k) {
      Frag16 a0, a1;
      load_frag(a0, &Alds[k][mi * 32 + lanelo][0], hi);
      load_frag(a1, &Alds[k][mi * 32 + 16 + lanelo][0], hi);
#pragma unroll
      for (int t = 0; t < 2; ++t) {
        Frag16 bfr;
        if (mbits[t] & (1u << k)) {
          load_frag(bfr, &Blds[npair * 32 + t * 16 + lanelo + k][0], hi);
        } else {
#pragma unroll
          for (int jj = 0; jj < 8; ++jj) bfr.u[jj] = 0u;
        }
        acc[0][t] = wmma_bf16(a0, bfr, acc[0][t]);
        acc[1][t] = wmma_bf16(a1, bfr, acc[1][t]);
      }
    }
    __syncthreads();
  }

  // epilogue: outputs are [B, L, C] -> 8 consecutive cout per lane => b128 stores
#pragma unroll
  for (int m = 0; m < 2; ++m) {
    const int row = c0 + mi * 32 + m * 16 + hi * 8;
    float bv[8];
#pragma unroll
    for (int r = 0; r < 8; ++r) bv[r] = bias[row + r];
#pragma unroll
    for (int t = 0; t < 2; ++t) {
      const int col = i0 + npair * 32 + t * 16 + lanelo;
      const size_t base = (size_t)(b * Ln + col) * Cn + row;
      if (LAST) {
        float4 f0, f1;
        f0.x = acc[m][t][0] + bv[0]; f0.y = acc[m][t][1] + bv[1];
        f0.z = acc[m][t][2] + bv[2]; f0.w = acc[m][t][3] + bv[3];
        f1.x = acc[m][t][4] + bv[4]; f1.y = acc[m][t][5] + bv[5];
        f1.z = acc[m][t][6] + bv[6]; f1.w = acc[m][t][7] + bv[7];
        *(float4*)(Yf + base) = f0;
        *(float4*)(Yf + base + 4) = f1;
      } else {
        Pack8 o;
#pragma unroll
        for (int r = 0; r < 8; ++r) o.s[r] = f2bf(acc[m][t][r] + bv[r]);
        *(uint4*)(Ybf + base) = o.v;
      }
    }
  }
}

// ---------------- layernorm (wave per row) ----------------

__global__ __launch_bounds__(256)
void ln_kernel(const float* __restrict__ a, const float* __restrict__ b2,
               const float* __restrict__ g, const float* __restrict__ bb,
               float* __restrict__ outf, ushort_t* __restrict__ outb) {
  const int row = blockIdx.x * 8 + (threadIdx.x >> 5);
  const int lane = threadIdx.x & 31;
  const size_t base = (size_t)row * Cn;
  float vals[16];
  float s = 0.f;
#pragma unroll
  for (int j = 0; j < 16; ++j) {
    int i = lane + j * 32;
    float v = a[base + i];
    if (b2) v += b2[base + i];
    vals[j] = v;
    s += v;
  }
#pragma unroll
  for (int off = 16; off > 0; off >>= 1) s += __shfl_xor(s, off, 32);
  float mean = s * (1.f / Cn);
  float q = 0.f;
#pragma unroll
  for (int j = 0; j < 16; ++j) { float d = vals[j] - mean; q += d * d; }
#pragma unroll
  for (int off = 16; off > 0; off >>= 1) q += __shfl_xor(q, off, 32);
  float rs = rsqrtf(q * (1.f / Cn) + EPSV);
#pragma unroll
  for (int j = 0; j < 16; ++j) {
    int i = lane + j * 32;
    float o = (vals[j] - mean) * rs * g[i] + bb[i];
    if (outf) outf[base + i] = o;
    if (outb) outb[base + i] = f2bf(o);
  }
}

// ---------------- MLP GEMMs (double-buffered LDS, async staging) ----------------

template <int EPI>
__global__ __launch_bounds__(256)
void gemm_kernel(const ushort_t* __restrict__ A, const ushort_t* __restrict__ Bt,
                 const float* __restrict__ bias, const float* __restrict__ resid,
                 void* __restrict__ outp, int N, int Kd) {
  constexpr int PITCH = 40;  // 80B row pitch: keeps 16B alignment, spreads banks
  __shared__ alignas(16) ushort_t Alds[2][128][PITCH];
  __shared__ alignas(16) ushort_t Blds[2][64][PITCH];
  const int n0 = blockIdx.x * 64;
  const int m0 = blockIdx.y * 128;
  const int tid = threadIdx.x;
  const int lane = tid & 31, lanelo = lane & 15, hi = lane >> 4;
  const int wave = tid >> 5, mi = wave & 3, npair = wave >> 2;

  const int aq = tid & 3, ar = tid >> 2;  // A: rows ar, ar+64
  const int bq = tid & 3, bn = tid >> 2;  // B: row bn
  const ushort_t* aptr0 = A + (size_t)(m0 + ar) * Kd + aq * 8;
  const ushort_t* aptr1 = aptr0 + (size_t)64 * Kd;
  const ushort_t* bptr  = Bt + (size_t)(n0 + bn) * Kd + bq * 8;

  v8f acc[2][2];
#pragma unroll
  for (int m = 0; m < 2; ++m)
#pragma unroll
    for (int t = 0; t < 2; ++t)
#pragma unroll
      for (int r = 0; r < 8; ++r) acc[m][t][r] = 0.f;

  const int nk = Kd / 32;
#if USE_ASYNC_LDS
  async_cp16(&Alds[0][ar][aq * 8], aptr0);
  async_cp16(&Alds[0][ar + 64][aq * 8], aptr1);
  async_cp16(&Blds[0][bn][bq * 8], bptr);
  aptr0 += 32; aptr1 += 32; bptr += 32;
  async_wait0();
  __syncthreads();
  for (int kb = 0; kb < nk; ++kb) {
    const int cur = kb & 1, nxt = cur ^ 1;
    if (kb + 1 < nk) {  // overlap next tile's async copies with WMMAs
      async_cp16(&Alds[nxt][ar][aq * 8], aptr0);
      async_cp16(&Alds[nxt][ar + 64][aq * 8], aptr1);
      async_cp16(&Blds[nxt][bn][bq * 8], bptr);
      aptr0 += 32; aptr1 += 32; bptr += 32;
    }
    Frag16 a0, a1, b0, b1;
    load_frag(a0, &Alds[cur][mi * 32 + lanelo][0], hi);
    load_frag(a1, &Alds[cur][mi * 32 + 16 + lanelo][0], hi);
    load_frag(b0, &Blds[cur][npair * 32 + lanelo][0], hi);
    load_frag(b1, &Blds[cur][npair * 32 + 16 + lanelo][0], hi);
    acc[0][0] = wmma_bf16(a0, b0, acc[0][0]);
    acc[0][1] = wmma_bf16(a0, b1, acc[0][1]);
    acc[1][0] = wmma_bf16(a1, b0, acc[1][0]);
    acc[1][1] = wmma_bf16(a1, b1, acc[1][1]);
    async_wait0();
    __syncthreads();
  }
#else
  uint4 ra0 = *(const uint4*)aptr0;
  uint4 ra1 = *(const uint4*)aptr1;
  uint4 rb  = *(const uint4*)bptr;
  aptr0 += 32; aptr1 += 32; bptr += 32;
  *(uint4*)&Alds[0][ar][aq * 8] = ra0;
  *(uint4*)&Alds[0][ar + 64][aq * 8] = ra1;
  *(uint4*)&Blds[0][bn][bq * 8] = rb;
  __syncthreads();
  for (int kb = 0; kb < nk; ++kb) {
    const int cur = kb & 1, nxt = cur ^ 1;
    const bool more = (kb + 1) < nk;
    if (more) {
      ra0 = *(const uint4*)aptr0;
      ra1 = *(const uint4*)aptr1;
      rb  = *(const uint4*)bptr;
      aptr0 += 32; aptr1 += 32; bptr += 32;
    }
    Frag16 a0, a1, b0, b1;
    load_frag(a0, &Alds[cur][mi * 32 + lanelo][0], hi);
    load_frag(a1, &Alds[cur][mi * 32 + 16 + lanelo][0], hi);
    load_frag(b0, &Blds[cur][npair * 32 + lanelo][0], hi);
    load_frag(b1, &Blds[cur][npair * 32 + 16 + lanelo][0], hi);
    acc[0][0] = wmma_bf16(a0, b0, acc[0][0]);
    acc[0][1] = wmma_bf16(a0, b1, acc[0][1]);
    acc[1][0] = wmma_bf16(a1, b0, acc[1][0]);
    acc[1][1] = wmma_bf16(a1, b1, acc[1][1]);
    if (more) {
      *(uint4*)&Alds[nxt][ar][aq * 8] = ra0;
      *(uint4*)&Alds[nxt][ar + 64][aq * 8] = ra1;
      *(uint4*)&Blds[nxt][bn][bq * 8] = rb;
    }
    __syncthreads();
  }
#endif

  if constexpr (EPI == 0) {
    // wave-private 32x32 LDS transpose -> contiguous b128 stores of G rows
    __shared__ alignas(16) ushort_t Tlds[8][32][PITCH];
    ushort_t* tw = &Tlds[wave][0][0];
#pragma unroll
    for (int t = 0; t < 2; ++t) {
      const int cl = t * 16 + lanelo;
      const float bv = bias[n0 + npair * 32 + cl];
#pragma unroll
      for (int m = 0; m < 2; ++m)
#pragma unroll
        for (int r = 0; r < 8; ++r) {
          const int rl = m * 16 + hi * 8 + r;
          float v = acc[m][t][r] + bv;
          float gl = 0.5f * v * (1.f + erff(v * 0.70710678118654752f));
          tw[rl * PITCH + cl] = f2bf(gl);
        }
    }
    const ushort_t* rbase = tw + lane * PITCH;
    uint4 o0 = *(const uint4*)(rbase);
    uint4 o1 = *(const uint4*)(rbase + 8);
    uint4 o2 = *(const uint4*)(rbase + 16);
    uint4 o3 = *(const uint4*)(rbase + 24);
    ushort_t* gdst = (ushort_t*)outp + (size_t)(m0 + mi * 32 + lane) * N + n0 + npair * 32;
    *(uint4*)(gdst) = o0;
    *(uint4*)(gdst + 8) = o1;
    *(uint4*)(gdst + 16) = o2;
    *(uint4*)(gdst + 24) = o3;
  } else {
#pragma unroll
    for (int t = 0; t < 2; ++t) {
      const int col = n0 + npair * 32 + t * 16 + lanelo;
      const float bv = bias[col];
#pragma unroll
      for (int m = 0; m < 2; ++m)
#pragma unroll
        for (int r = 0; r < 8; ++r) {
          const int row = m0 + mi * 32 + m * 16 + hi * 8 + r;
          float v = acc[m][t][r] + bv + resid[(size_t)row * N + col];
          ((float*)outp)[(size_t)row * N + col] = v;
        }
    }
  }
}

}  // namespace

extern "C" void kernel_launch(void* const* d_in, const int* in_sizes, int n_in,
                              void* d_out, int out_size, void* d_ws, size_t ws_size,
                              hipStream_t stream) {
  (void)in_sizes; (void)n_in; (void)out_size; (void)ws_size;
  const float* x     = (const float*)d_in[0];
  const int*   ids   = (const int*)d_in[1];
  const float* cw[3] = {(const float*)d_in[2], (const float*)d_in[4], (const float*)d_in[6]};
  const float* cbp[3]= {(const float*)d_in[3], (const float*)d_in[5], (const float*)d_in[7]};
  const float* w_in  = (const float*)d_in[8];
  const float* b_in  = (const float*)d_in[9];
  const float* w_out = (const float*)d_in[10];
  const float* b_out = (const float*)d_in[11];
  const float* ln1g  = (const float*)d_in[12];
  const float* ln1b  = (const float*)d_in[13];
  const float* ln2g  = (const float*)d_in[14];
  const float* ln2b  = (const float*)d_in[15];

  char* ws = (char*)d_ws;
  const size_t MB = 1ull << 20;
  // Conv phase region [0, 128MB): Xbf(32) | Ybf(32) | Y3 f32(64). Reused as G (128MB bf16).
  ushort_t* Xbf   = (ushort_t*)(ws + 0);          // [B,L,C] bf16
  ushort_t* Ybf   = (ushort_t*)(ws + 32 * MB);    // [B,L,C] bf16
  float*    Y3    = (float*)(ws + 64 * MB);       // [B,L,C] f32
  ushort_t* G     = (ushort_t*)(ws + 0);
  float*    h     = (float*)(ws + 128 * MB);
  ushort_t* hbf   = (ushort_t*)(ws + 192 * MB);
  float*    tbuf  = (float*)(ws + 224 * MB);
  ushort_t* Wc    = (ushort_t*)(ws + 288 * MB);   // 15 taps * 512*512 bf16 = 7.5MB
  ushort_t* Winb  = (ushort_t*)(ws + 297 * MB);   // W_in^T  [H][C] bf16, 2MB
  ushort_t* Woutb = (ushort_t*)(ws + 300 * MB);   // W_out^T [C][H] bf16, 2MB

  // ---- prep ----
  cvt_bf16_kernel<<<(BLn * Cn / 8 + 255) / 256, 256, 0, stream>>>(x, Xbf, BLn * Cn / 8);
  const int KS[3] = {3, 5, 7};
  const size_t wcoff[3] = {0, (size_t)3 * Cn * Cn, (size_t)8 * Cn * Cn};
  for (int i = 0; i < 3; ++i) {
    int total = Cn * Cn * KS[i];
    conv_w_reorder_kernel<<<(total + 255) / 256, 256, 0, stream>>>(cw[i], Wc + wcoff[i],
                                                                   KS[i], total);
  }
  wtrans_kernel<<<dim3(Hn / 32, Cn / 32), dim3(32, 8), 0, stream>>>(w_in, Winb, Cn, Hn);
  wtrans_kernel<<<dim3(Cn / 32, Hn / 32), dim3(32, 8), 0, stream>>>(w_out, Woutb, Hn, Cn);

  // ---- masked conv chain (WMMA GEMMs), activations stay [B,L,C] ----
  dim3 cgrid(Ln / 64, Cn / 128, Bn);
  conv_kernel<3, false><<<cgrid, 256, 0, stream>>>(Xbf, Wc + wcoff[0], cbp[0], ids, Ybf, nullptr);
  conv_kernel<5, false><<<cgrid, 256, 0, stream>>>(Ybf, Wc + wcoff[1], cbp[1], ids, Xbf, nullptr);
  conv_kernel<7, true ><<<cgrid, 256, 0, stream>>>(Xbf, Wc + wcoff[2], cbp[2], ids, nullptr, Y3);

  // ---- LN1: h = LN(x + conv_out) ----
  ln_kernel<<<BLn / 8, 256, 0, stream>>>(x, Y3, ln1g, ln1b, h, hbf);

  // ---- MLP (WMMA GEMMs) + residual, then LN2 ----
  gemm_kernel<0><<<dim3(Hn / 64, BLn / 128), 256, 0, stream>>>(hbf, Winb, b_in, nullptr,
                                                               (void*)G, Hn, Cn);
  gemm_kernel<1><<<dim3(Cn / 64, BLn / 128), 256, 0, stream>>>(G, Woutb, b_out, h,
                                                               (void*)tbuf, Cn, Hn);
  ln_kernel<<<BLn / 8, 256, 0, stream>>>(tbuf, nullptr, ln2g, ln2b, (float*)d_out, nullptr);
}
